// Head_40080634806272
// MI455X (gfx1250) — matrix operand order repaired and checked by reading, not compiled, not run on hardware
//
#include <hip/hip_runtime.h>

// ---------------------------------------------------------------------------
// CDNA5 / gfx1250 implementation.
//   Stage 1: Q/K/V projections as split-bf16 WMMA GEMMs (f32-accurate):
//            X = Xhi + Xlo, W = Whi + Wlo (bf16 each);
//            X@W ~= Xhi@Whi + Xhi@Wlo + Xlo@Whi  via v_wmma_f32_16x16x32_bf16
//            Register-staged pipeline: global loads of tile k+1 overlap the
//            WMMAs of tile k.
//   Stage 2: per-row rank-1 attention softmax (attn row = softmax(q_i*s * k))
// ---------------------------------------------------------------------------

typedef __attribute__((ext_vector_type(4)))  __bf16 v4bf;
typedef __attribute__((ext_vector_type(8)))  __bf16 v8bf;
typedef __attribute__((ext_vector_type(16))) __bf16 v16bf;
typedef __attribute__((ext_vector_type(8)))  float  v8f;

#define BATCH  16384
#define NEMBD  1568
#define HEAD   128
#define BK     32
#define KSTEPS (NEMBD / BK)   // 49 (exact: 49*32 = 1568)
#define LSTR   40             // LDS row stride in bf16 elems (32 + 8 pad)

#define SCALE  0.02525381361380526f   // 1/sqrt(1568)

union BF16x16 { v16bf v; v8bf h[2]; };

// A fragment (16-bit A 16x32 layout): lane<16 -> K {0..7,16..23},
// lane>=16 -> K {8..15,24..31}; two contiguous 16B chunks per lane.
__device__ __forceinline__ v16bf frag_a(const __bf16* rowbase, int sel) {
  BF16x16 u;
  u.h[0] = *(const v8bf*)(rowbase + sel * 8);
  u.h[1] = *(const v8bf*)(rowbase + 16 + sel * 8);
  return u.v;
}

// B fragment: lane holds column n = lane%16, K = (lane/16)*16 .. +15,
// contiguous in K (LDS holds B transposed: [n][k]).
__device__ __forceinline__ v16bf frag_b(const __bf16* p) {
  BF16x16 u;
  u.h[0] = *(const v8bf*)(p);
  u.h[1] = *(const v8bf*)(p + 8);
  return u.v;
}

__global__ __launch_bounds__(256) void proj_gemm_kernel(
    const float* __restrict__ x,
    const float* __restrict__ wq,
    const float* __restrict__ wk,
    const float* __restrict__ wv,
    float* __restrict__ qkv)
{
  __shared__ __bf16 Ah[128 * LSTR];
  __shared__ __bf16 Al[128 * LSTR];
  __shared__ __bf16 Bh[128 * LSTR];   // transposed: [n][k]
  __shared__ __bf16 Bl[128 * LSTR];

  const int tid  = threadIdx.x;          // 0..255 (8 waves, wave32)
  const int m0   = blockIdx.x * 128;     // batch-row base
  const int wsel = blockIdx.y;           // which weight matrix
  const float* __restrict__ w = (wsel == 0) ? wq : ((wsel == 1) ? wk : wv);
  float* __restrict__ out = qkv + (size_t)wsel * (size_t)BATCH * HEAD;

  const int wave = tid >> 5;
  const int lane = tid & 31;
  const int wm   = wave >> 2;            // 0..1 : 64-row strip
  const int wn   = wave & 3;             // 0..3 : 32-col strip
  const int r    = lane & 15;
  const int sel  = lane >> 4;

  // loader indices (256 threads move the whole 128x32 A / 32x128 B tile)
  const int arow0 = tid >> 3;            // + {0,32,64,96}
  const int acol  = (tid & 7) * 4;
  const int brow0 = tid >> 5;            // + {0,8,16,24}
  const int bn    = (tid & 31) * 4;

  v8f acc[4][2];
#pragma unroll
  for (int mt = 0; mt < 4; ++mt)
#pragma unroll
    for (int nt = 0; nt < 2; ++nt) {
      v8f z = {0.f, 0.f, 0.f, 0.f, 0.f, 0.f, 0.f, 0.f};
      acc[mt][nt] = z;
    }

  // register staging buffers (raw f32 tile in flight)
  float4 ra[4], rb[4];

  auto load_gl = [&](int k0) {
#pragma unroll
    for (int it = 0; it < 4; ++it) {
      const int arow = arow0 + 32 * it;
      ra[it] = *(const float4*)(x + (size_t)(m0 + arow) * NEMBD + k0 + acol);
    }
#pragma unroll
    for (int it = 0; it < 4; ++it) {
      const int brow = brow0 + 8 * it;
      rb[it] = *(const float4*)(w + (size_t)(k0 + brow) * HEAD + bn);
    }
  };

  load_gl(0);   // prologue: tile 0 into registers

  for (int kk = 0; kk < KSTEPS; ++kk) {
    // ---- stage registers -> LDS, splitting f32 into bf16 hi/lo ----
#pragma unroll
    for (int it = 0; it < 4; ++it) {
      const int arow = arow0 + 32 * it;
      const float4 f = ra[it];
      const __bf16 h0 = (__bf16)f.x, h1 = (__bf16)f.y,
                   h2 = (__bf16)f.z, h3 = (__bf16)f.w;
      const v4bf hv = {h0, h1, h2, h3};
      const v4bf lv = {(__bf16)(f.x - (float)h0), (__bf16)(f.y - (float)h1),
                       (__bf16)(f.z - (float)h2), (__bf16)(f.w - (float)h3)};
      *(v4bf*)(Ah + arow * LSTR + acol) = hv;
      *(v4bf*)(Al + arow * LSTR + acol) = lv;
    }
#pragma unroll
    for (int it = 0; it < 4; ++it) {
      const int brow = brow0 + 8 * it;
      const float4 f = rb[it];
      const float fv[4] = {f.x, f.y, f.z, f.w};
#pragma unroll
      for (int c = 0; c < 4; ++c) {      // scatter transposed: [n][k]
        const __bf16 h = (__bf16)fv[c];
        Bh[(bn + c) * LSTR + brow] = h;
        Bl[(bn + c) * LSTR + brow] = (__bf16)(fv[c] - (float)h);
      }
    }
    __syncthreads();

    // ---- prefetch next tile: overlaps with fragment loads + WMMAs ----
    if (kk + 1 < KSTEPS) load_gl((kk + 1) * BK);

    // ---- LDS -> fragments ----
    v16bf ahi[4], alo[4], bhi[2], blo[2];
#pragma unroll
    for (int mt = 0; mt < 4; ++mt) {
      const int row = wm * 64 + mt * 16 + r;
      ahi[mt] = frag_a(Ah + row * LSTR, sel);
      alo[mt] = frag_a(Al + row * LSTR, sel);
    }
#pragma unroll
    for (int nt = 0; nt < 2; ++nt) {
      const int n = wn * 32 + nt * 16 + r;
      bhi[nt] = frag_a(Bh + n * LSTR, sel);
      blo[nt] = frag_a(Bl + n * LSTR, sel);
    }

    // ---- split-bf16 WMMA: hi*hi + hi*lo + lo*hi ----
#pragma unroll
    for (int mt = 0; mt < 4; ++mt)
#pragma unroll
      for (int nt = 0; nt < 2; ++nt) {
        v8f c = acc[mt][nt];
        c = __builtin_amdgcn_wmma_f32_16x16x32_bf16(
            false, alo[mt], false, bhi[nt], (short)0, c, false, false);
        c = __builtin_amdgcn_wmma_f32_16x16x32_bf16(
            false, ahi[mt], false, blo[nt], (short)0, c, false, false);
        c = __builtin_amdgcn_wmma_f32_16x16x32_bf16(
            false, ahi[mt], false, bhi[nt], (short)0, c, false, false);
        acc[mt][nt] = c;
      }
    __syncthreads();
  }

  // ---- writeback (C layout: VGPR e -> row e + sel*8, col = lane%16) ----
#pragma unroll
  for (int mt = 0; mt < 4; ++mt)
#pragma unroll
    for (int nt = 0; nt < 2; ++nt) {
      const int col = wn * 32 + nt * 16 + r;
#pragma unroll
      for (int e = 0; e < 8; ++e) {
        const int row = m0 + wm * 64 + mt * 16 + e + sel * 8;
        out[(size_t)row * HEAD + col] = acc[mt][nt][e];
      }
    }
}

// ---------------------------------------------------------------------------
// Stage 2: attn[b,i,:] = softmax(q[b,i]*scale * k[b,:]);  out = attn @ v[b,:]
// One 128-thread block per batch row; thread i produces out[b,i].
// ---------------------------------------------------------------------------
__global__ __launch_bounds__(128) void attn_softmax_kernel(
    const float* __restrict__ q, const float* __restrict__ k,
    const float* __restrict__ v, float* __restrict__ out)
{
  __shared__ float ks[HEAD];
  __shared__ float vs[HEAD];
  const int b = blockIdx.x;
  const int i = threadIdx.x;
  const size_t base = (size_t)b * HEAD;

  ks[i] = k[base + i];
  vs[i] = v[base + i];
  __syncthreads();

  const float c = q[base + i] * SCALE;

  float m = -3.402823466e38f;
#pragma unroll 8
  for (int j = 0; j < HEAD; ++j) m = fmaxf(m, c * ks[j]);

  float s = 0.f, a = 0.f;
#pragma unroll 8
  for (int j = 0; j < HEAD; ++j) {
    const float e = __expf(c * ks[j] - m);
    s += e;
    a += e * vs[j];
  }
  out[base + i] = a / s;
}

extern "C" void kernel_launch(void* const* d_in, const int* in_sizes, int n_in,
                              void* d_out, int out_size, void* d_ws, size_t ws_size,
                              hipStream_t stream) {
  (void)in_sizes; (void)n_in; (void)out_size; (void)ws_size;
  const float* x  = (const float*)d_in[0];
  const float* wq = (const float*)d_in[1];
  const float* wk = (const float*)d_in[2];
  const float* wv = (const float*)d_in[3];
  float* qkv = (float*)d_ws;     // [3][16384][128] f32 scratch (25.2 MB)
  float* out = (float*)d_out;

  proj_gemm_kernel<<<dim3(BATCH / 128, 3), 256, 0, stream>>>(x, wq, wk, wv, qkv);
  attn_softmax_kernel<<<BATCH, 128, 0, stream>>>(
      qkv, qkv + (size_t)BATCH * HEAD, qkv + (size_t)2 * BATCH * HEAD, out);
}
